// MultiHeadAttentionV1_44753559224991
// MI455X (gfx1250) — matrix operand order repaired and checked by reading, not compiled, not run on hardware
//
#include <hip/hip_runtime.h>
#include <hip/hip_bf16.h>

typedef __attribute__((ext_vector_type(16))) __bf16 v16bf;
typedef __attribute__((ext_vector_type(8)))  float  v8f;

#define WMMA_BF16(a, b, c) \
  __builtin_amdgcn_wmma_f32_16x16x32_bf16(false, (a), false, (b), (short)0, (c), false, false)

// ---- fragment loaders (wave32 layouts per CDNA5 ISA 7.12.2) ----
// A 16x32 bf16 from row-major LDS tile: row = lane&15 ; K = 8*(lane>>4)+{0..7}, then +16
__device__ __forceinline__ v16bf lds_load_A(const __bf16* p, int ld) {
  const int lane = threadIdx.x & 31;
  const __bf16* r = p + (lane & 15) * ld + ((lane >> 4) << 3);
  v16bf a;
#pragma unroll
  for (int i = 0; i < 8; ++i) { a[i] = r[i]; a[i + 8] = r[i + 16]; }
  return a;
}
// B 32x16 bf16 from B^T-layout LDS tile [n][k]: col = lane&15 ; K = 8*(lane>>4)+{0..7}, then +16
__device__ __forceinline__ v16bf lds_load_B(const __bf16* p, int ld) {
  const int lane = threadIdx.x & 31;
  const __bf16* r = p + (lane & 15) * ld + ((lane >> 4) << 3);
  v16bf b;
#pragma unroll
  for (int i = 0; i < 8; ++i) { b[i] = r[i]; b[i + 8] = r[i + 16]; }
  return b;
}

// Async global->LDS stage: 32B at aLds<-aG, 64B at bLds<-bG (b128 chunks).
// INST_OFFSET applies to BOTH LDS and global addresses (ISA 10.4.4), so two
// base addresses cover all six transfers. Tracked by ASYNCcnt.
__device__ __forceinline__ void async_stage(unsigned aLds, unsigned bLds,
                                            const __bf16* aG, const __bf16* bG) {
  asm volatile(
      "global_load_async_to_lds_b128 %0, %2, off\n\t"
      "global_load_async_to_lds_b128 %0, %2, off offset:16\n\t"
      "global_load_async_to_lds_b128 %1, %3, off\n\t"
      "global_load_async_to_lds_b128 %1, %3, off offset:16\n\t"
      "global_load_async_to_lds_b128 %1, %3, off offset:32\n\t"
      "global_load_async_to_lds_b128 %1, %3, off offset:48"
      :: "v"(aLds), "v"(bLds), "v"(aG), "v"(bG)
      : "memory");
}
#define ASYNC_WAIT() asm volatile("s_wait_asynccnt 0" ::: "memory")

// =====================================================================
// Pre-pass kernels: cast x -> bf16; build B^T bf16 operands so the GEMM
// hot loops are pure async b128 copies (no per-element cvt VALU).
// =====================================================================
__global__ __launch_bounds__(256) void cast_x_kernel(const float* __restrict__ x,
                                                     __bf16* __restrict__ xb) {
  const size_t i0 = ((size_t)blockIdx.x * 256 + threadIdx.x) * 8;
#pragma unroll
  for (int i = 0; i < 8; ++i) xb[i0 + i] = (__bf16)x[i0 + i];
}

// bqkvt[c][d] = wqkv[c>>10][ (c&1023)>>6 ][d][ c&63 ], c in [0,3072)
__global__ __launch_bounds__(256) void build_bqkvt_kernel(
    const float* __restrict__ wq, const float* __restrict__ wk,
    const float* __restrict__ wv, __bf16* __restrict__ bt) {
  const size_t idx = (size_t)blockIdx.x * 256 + threadIdx.x;
  const int c = (int)(idx >> 10), d = (int)(idx & 1023);
  const int mat = c >> 10, hc = c & 1023, h = hc >> 6, kk = hc & 63;
  const float* wp = (mat == 0) ? wq : (mat == 1) ? wk : wv;
  bt[idx] = (__bf16)wp[(size_t)h * 65536 + (size_t)d * 64 + kk];
}

// wt[d][hk] = w[hk][d]
__global__ __launch_bounds__(256) void build_wt_kernel(const float* __restrict__ w,
                                                       __bf16* __restrict__ wt) {
  const size_t idx = (size_t)blockIdx.x * 256 + threadIdx.x;
  const int d = (int)(idx >> 10), hk = (int)(idx & 1023);
  wt[idx] = (__bf16)w[(size_t)hk * 1024 + d];
}

// =====================================================================
// GEMM tiling (kernels 1 & 3): block 128x256, BK=32; 8 waves (2x4),
// wave tile 64x64 = 16 WMMAs/step. Double-buffered LDS; next tile staged
// via GLOBAL_LOAD_ASYNC_TO_LDS_B128 (no VGPR staging -> no spills),
// overlapping the WMMAs; s_wait_asynccnt 0 before the barrier.
// =====================================================================
#define GEMM_BUF_ELEMS (128 * 48 + 256 * 48)   // 18432 bf16 elems per buffer
#define GEMM_LDS_BYTES (2 * GEMM_BUF_ELEMS * 2)

__global__ __launch_bounds__(256, 1) void qkv_gemm_kernel(
    const __bf16* __restrict__ xb, const __bf16* __restrict__ bt,
    __bf16* __restrict__ qo, __bf16* __restrict__ ko, __bf16* __restrict__ vo) {
  extern __shared__ __bf16 gsm[];
  const int tid = threadIdx.x, lane = tid & 31, wv_id = tid >> 5;
  const int hi = lane >> 4, ln = lane & 15;
  const int m0 = blockIdx.y * 128, c0 = blockIdx.x * 256;
  const int wm = wv_id >> 2, wn = wv_id & 3;

  const int arow = tid >> 1, acb = (tid & 1) * 16;
  const __bf16* asrc = xb + (size_t)(m0 + arow) * 1024 + acb;
  const __bf16* bsrc = bt + (size_t)(c0 + tid) * 1024;
  const int a_lds = arow * 48 + acb;   // element offsets within a buffer
  const int b_lds = 128 * 48 + tid * 48;

  // prologue: stage tile 0 into buffer 0 (async DMA)
  async_stage((unsigned)(size_t)(gsm + a_lds), (unsigned)(size_t)(gsm + b_lds),
              asrc, bsrc);
  ASYNC_WAIT();
  __syncthreads();

  v8f acc[4][4];
#pragma unroll
  for (int i = 0; i < 4; ++i)
#pragma unroll
    for (int j = 0; j < 4; ++j)
#pragma unroll
      for (int r = 0; r < 8; ++r) acc[i][j][r] = 0.f;

  int p = 0;
  for (int kt = 0; kt < 1024; kt += 32) {
    const bool pre = (kt + 32) < 1024;
    if (pre) {   // async-stage next tile into buf p^1; overlaps WMMAs below
      __bf16* An = gsm + (p ^ 1) * GEMM_BUF_ELEMS;
      async_stage((unsigned)(size_t)(An + a_lds), (unsigned)(size_t)(An + b_lds),
                  asrc + kt + 32, bsrc + kt + 32);
    }
    const __bf16* Ab = gsm + p * GEMM_BUF_ELEMS;
    const __bf16* Bb = Ab + 128 * 48;
    v16bf bfr[4];
#pragma unroll
    for (int j = 0; j < 4; ++j) bfr[j] = lds_load_B(Bb + (wn * 64 + j * 16) * 48, 48);
#pragma unroll
    for (int i = 0; i < 4; ++i) {
      v16bf af = lds_load_A(Ab + (wm * 64 + i * 16) * 48, 48);
#pragma unroll
      for (int j = 0; j < 4; ++j) acc[i][j] = WMMA_BF16(af, bfr[j], acc[i][j]);
    }
    ASYNC_WAIT();
    __syncthreads();
    p ^= 1;
  }

  // epilogue -> q/k/v bf16, [B*S, H*K] layout
#pragma unroll
  for (int j = 0; j < 4; ++j) {
    const int c = c0 + wn * 64 + j * 16 + ln;
    const int mat = c >> 10, cc = c & 1023;
    __bf16* outp = (mat == 0) ? qo : (mat == 1) ? ko : vo;
#pragma unroll
    for (int i = 0; i < 4; ++i) {
      const int rbase = m0 + wm * 64 + i * 16 + hi * 8;
#pragma unroll
      for (int r = 0; r < 8; ++r)
        outp[(size_t)(rbase + r) * 1024 + cc] = (__bf16)acc[i][j][r];
    }
  }
}

// =====================================================================
// Kernel 2: flash attention per (b,h). 128 query rows/block, key tiles
// of 128. Dynamic LDS (~87KB). Each wave owns 16 query rows.
// =====================================================================
__global__ __launch_bounds__(256) void flash_attn_kernel(
    const __bf16* __restrict__ qws, const __bf16* __restrict__ kws,
    const __bf16* __restrict__ vws, __bf16* __restrict__ hws) {
  extern __shared__ __bf16 smem[];
  __bf16* Qs = smem;                    // [128][72]
  __bf16* Ks = Qs + 128 * 72;           // [128][72]   (== B^T for scores)
  __bf16* Vt = Ks + 128 * 72;           // [64][136]   (== B^T for attn*V)
  __bf16* Ps = Vt + 64 * 136;           // 8 x [16][136] per-wave P staging

  const int tid = threadIdx.x, lane = tid & 31, wv_id = tid >> 5;
  const int hi = lane >> 4, ln = lane & 15;
  const int b = blockIdx.y >> 4, h = blockIdx.y & 15;
  const int s0 = blockIdx.x * 128;
  const size_t rowbase = (size_t)b * 2048;
  __bf16* Pw = Ps + wv_id * (16 * 136);

  { // load Q once, fold in softmax scale 1/8 (exact in bf16)
    const int row = tid >> 1, cb = (tid & 1) * 32;
    const __bf16* src = qws + (rowbase + s0 + row) * 1024 + h * 64 + cb;
#pragma unroll
    for (int i = 0; i < 32; ++i) Qs[row * 72 + cb + i] = (__bf16)(0.125f * (float)src[i]);
  }

  float m_run[8], l_run[8];
  v8f acc[4];
#pragma unroll
  for (int r = 0; r < 8; ++r) { m_run[r] = -1e30f; l_run[r] = 0.f; }
#pragma unroll
  for (int n = 0; n < 4; ++n)
#pragma unroll
    for (int r = 0; r < 8; ++r) acc[n][r] = 0.f;

  for (int t0 = 0; t0 < 2048; t0 += 128) {
    __syncthreads();
    { // stage K [t][k] via b128 copies
      const int t = tid >> 1, cb = (tid & 1) * 32;
      const uint4* src = (const uint4*)(kws + (rowbase + t0 + t) * 1024 + h * 64 + cb);
      uint4* dst = (uint4*)(Ks + t * 72 + cb);
#pragma unroll
      for (int i = 0; i < 4; ++i) dst[i] = src[i];
    }
    { // stage V transposed [k][t]
      const int t = tid >> 1, kb = (tid & 1) * 32;
      const __bf16* src = vws + (rowbase + t0 + t) * 1024 + h * 64 + kb;
#pragma unroll
      for (int i = 0; i < 32; ++i) Vt[(kb + i) * 136 + t] = src[i];
    }
    __syncthreads();

    // scores: 16 rows x 128 cols, K=64 in two WMMA steps -> 16 WMMAs
    v8f sacc[8];
#pragma unroll
    for (int j = 0; j < 8; ++j)
#pragma unroll
      for (int r = 0; r < 8; ++r) sacc[j][r] = 0.f;
#pragma unroll
    for (int ks = 0; ks < 64; ks += 32) {
      v16bf a = lds_load_A(Qs + (wv_id * 16) * 72 + ks, 72);
#pragma unroll
      for (int j = 0; j < 8; ++j) {
        v16bf bb = lds_load_B(Ks + (j * 16) * 72 + ks, 72);
        sacc[j] = WMMA_BF16(a, bb, sacc[j]);
      }
    }

    // online softmax; C-layout element r <-> row r+8*hi, col ln.
#pragma unroll
    for (int r = 0; r < 8; ++r) {
      float tmax = sacc[0][r];
#pragma unroll
      for (int j = 1; j < 8; ++j) tmax = fmaxf(tmax, sacc[j][r]);
#pragma unroll
      for (int off = 1; off < 16; off <<= 1)
        tmax = fmaxf(tmax, __shfl_xor(tmax, off, 32));
      const float m_new = fmaxf(m_run[r], tmax);
      const float rescale = __expf(m_run[r] - m_new);
      float lsum = 0.f;
#pragma unroll
      for (int j = 0; j < 8; ++j) {
        const float p = __expf(sacc[j][r] - m_new);
        sacc[j][r] = p;
        lsum += p;
      }
#pragma unroll
      for (int off = 1; off < 16; off <<= 1)
        lsum += __shfl_xor(lsum, off, 32);
      l_run[r] = l_run[r] * rescale + lsum;
      m_run[r] = m_new;
#pragma unroll
      for (int n = 0; n < 4; ++n) acc[n][r] *= rescale;
    }

    // P (D-layout) -> per-wave LDS -> A-layout (wave-private; drain LDS only)
#pragma unroll
    for (int j = 0; j < 8; ++j)
#pragma unroll
      for (int r = 0; r < 8; ++r)
        Pw[(r + 8 * hi) * 136 + j * 16 + ln] = (__bf16)sacc[j][r];
    asm volatile("s_wait_dscnt 0" ::: "memory");

    // ctx += P @ V : 4 t-steps x 4 k-frags -> 16 WMMAs
#pragma unroll
    for (int ts = 0; ts < 128; ts += 32) {
      v16bf a = lds_load_A(Pw + ts, 136);
#pragma unroll
      for (int n = 0; n < 4; ++n) {
        v16bf bb = lds_load_B(Vt + (n * 16) * 136 + ts, 136);
        acc[n] = WMMA_BF16(a, bb, acc[n]);
      }
    }
  }

  // normalize and store heads (bf16) to [B*S, H*K]
#pragma unroll
  for (int r = 0; r < 8; ++r) {
    const float inv = 1.f / l_run[r];
    const size_t srow = rowbase + s0 + wv_id * 16 + r + 8 * hi;
#pragma unroll
    for (int n = 0; n < 4; ++n)
      hws[srow * 1024 + h * 64 + n * 16 + ln] = (__bf16)(acc[n][r] * inv);
  }
}

// =====================================================================
// Kernel 3: output projection. heads bf16 [8192x1024] x w^T bf16 -> f32.
// Same async double-buffered tiling as kernel 1.
// =====================================================================
__global__ __launch_bounds__(256, 1) void out_proj_kernel(
    const __bf16* __restrict__ hws, const __bf16* __restrict__ wt,
    float* __restrict__ out) {
  extern __shared__ __bf16 gsm[];
  const int tid = threadIdx.x, lane = tid & 31, wv_id = tid >> 5;
  const int hi = lane >> 4, ln = lane & 15;
  const int m0 = blockIdx.y * 128, c0 = blockIdx.x * 256;
  const int wm = wv_id >> 2, wn = wv_id & 3;

  const int arow = tid >> 1, acb = (tid & 1) * 16;
  const __bf16* asrc = hws + (size_t)(m0 + arow) * 1024 + acb;
  const __bf16* bsrc = wt + (size_t)(c0 + tid) * 1024;
  const int a_lds = arow * 48 + acb;
  const int b_lds = 128 * 48 + tid * 48;

  async_stage((unsigned)(size_t)(gsm + a_lds), (unsigned)(size_t)(gsm + b_lds),
              asrc, bsrc);
  ASYNC_WAIT();
  __syncthreads();

  v8f acc[4][4];
#pragma unroll
  for (int i = 0; i < 4; ++i)
#pragma unroll
    for (int j = 0; j < 4; ++j)
#pragma unroll
      for (int r = 0; r < 8; ++r) acc[i][j][r] = 0.f;

  int p = 0;
  for (int kt = 0; kt < 1024; kt += 32) {
    const bool pre = (kt + 32) < 1024;
    if (pre) {
      __bf16* An = gsm + (p ^ 1) * GEMM_BUF_ELEMS;
      async_stage((unsigned)(size_t)(An + a_lds), (unsigned)(size_t)(An + b_lds),
                  asrc + kt + 32, bsrc + kt + 32);
    }
    const __bf16* Ab = gsm + p * GEMM_BUF_ELEMS;
    const __bf16* Bb = Ab + 128 * 48;
    v16bf bfr[4];
#pragma unroll
    for (int j = 0; j < 4; ++j) bfr[j] = lds_load_B(Bb + (wn * 64 + j * 16) * 48, 48);
#pragma unroll
    for (int i = 0; i < 4; ++i) {
      v16bf af = lds_load_A(Ab + (wm * 64 + i * 16) * 48, 48);
#pragma unroll
      for (int j = 0; j < 4; ++j) acc[i][j] = WMMA_BF16(af, bfr[j], acc[i][j]);
    }
    ASYNC_WAIT();
    __syncthreads();
    p ^= 1;
  }

#pragma unroll
  for (int j = 0; j < 4; ++j) {
    const int c = c0 + wn * 64 + j * 16 + ln;
#pragma unroll
    for (int i = 0; i < 4; ++i) {
      const int rbase = m0 + wm * 64 + i * 16 + hi * 8;
#pragma unroll
      for (int r = 0; r < 8; ++r)
        out[(size_t)(rbase + r) * 1024 + c] = acc[i][j][r];
    }
  }
}

extern "C" void kernel_launch(void* const* d_in, const int* in_sizes, int n_in,
                              void* d_out, int out_size, void* d_ws, size_t ws_size,
                              hipStream_t stream) {
  const float* x  = (const float*)d_in[0];
  const float* wq = (const float*)d_in[1];
  const float* wk = (const float*)d_in[2];
  const float* wv = (const float*)d_in[3];
  const float* w  = (const float*)d_in[4];
  float* out = (float*)d_out;

  // scratch layout (bf16 elements)
  const size_t seg = (size_t)8192 * 1024;      // 8M elements
  __bf16* qws = (__bf16*)d_ws;                 // q               (16 MB)
  __bf16* kws = qws + seg;                     // k               (16 MB)
  __bf16* vws = kws + seg;                     // v               (16 MB)
  __bf16* hws = vws + seg;                     // concat heads    (16 MB)
  __bf16* xb  = hws + seg;                     // x in bf16       (16 MB)
  __bf16* bqt = xb + seg;                      // qkv weights^T   ( 6 MB)
  __bf16* wtt = bqt + (size_t)3072 * 1024;     // w^T             ( 2 MB)

  dim3 blk(256);
  cast_x_kernel<<<dim3(4096), blk, 0, stream>>>(x, xb);
  build_bqkvt_kernel<<<dim3(12288), blk, 0, stream>>>(wq, wk, wv, bqt);
  build_wt_kernel<<<dim3(4096), blk, 0, stream>>>(w, wtt);

  qkv_gemm_kernel<<<dim3(12, 64), blk, GEMM_LDS_BYTES, stream>>>(xb, bqt, qws, kws, vws);

  const int attn_lds = (128 * 72 + 128 * 72 + 64 * 136 + 8 * 16 * 136) * (int)sizeof(__bf16);
  flash_attn_kernel<<<dim3(16, 64), blk, attn_lds, stream>>>(qws, kws, vws, hws);

  out_proj_kernel<<<dim3(4, 64), blk, GEMM_LDS_BYTES, stream>>>(hws, wtt, out);
}
